// Mamba_70540542870057
// MI455X (gfx1250) — compile-verified, matrix operand order checked by reading, not run on hardware
//
#include <hip/hip_runtime.h>
#include <hip/hip_bf16.h>

// ---------------- problem constants (from reference) ----------------
constexpr int D_MODEL   = 1024;
constexpr int D_STATE   = 64;
constexpr int HEADDIM   = 64;
constexpr int D_INNER   = 2 * D_MODEL;                       // 2048
constexpr int NHEADS    = D_INNER / HEADDIM;                 // 32
constexpr int D_IN_PROJ = 2 * D_INNER + 2 * D_STATE + NHEADS;// 4256
constexpr int CHANNELS  = D_INNER + 2 * D_STATE;             // 2176
constexpr int KSIZE     = 4;
constexpr int BSZ       = 2;
constexpr int LSEQ      = 2048;

// ---------------- WMMA types ----------------
typedef __bf16 bf16_t;
typedef __attribute__((ext_vector_type(16))) __bf16 v16bf;
typedef __attribute__((ext_vector_type(4)))  __bf16 bf16x4;
typedef __attribute__((ext_vector_type(8)))  float  v8f;

// ---------------- bf16 WMMA GEMM: C[M,N] = A[M,K] @ W[N,K]^T ----------------
// Requirements (hold for every call site below):
//   M % 128 == 0, K % 32 == 0, all pointers/strides 16B aligned. Only N is
//   allowed ragged (guarded).
// Block: 256 threads = 8 waves, tile 128(M) x 128(N), BK=32.
// Waves arranged 4(M) x 2(N); each wave computes 32x64 via 2x4 WMMA accs.
// LDS ping-pong double buffering: global float4 loads for step s+1 are staged
// in VGPRs during the WMMAs of step s.
#define BM 128
#define BN 128
#define BK 32

__global__ __launch_bounds__(256) void gemm_bf16_wmma(
    const float* __restrict__ A, int lda,
    const float* __restrict__ W, int ldb,
    float* __restrict__ C, int ldc,
    int M, int N, int K, int accumulate, int epilogue)
{
    __shared__ alignas(16) bf16_t As[2][BM][BK];
    __shared__ alignas(16) bf16_t Bs[2][BN][BK];

    const int tid  = threadIdx.x;
    const int wave = tid >> 5;
    const int lane = tid & 31;
    const int half = lane >> 4;   // 0 or 1
    const int l15  = lane & 15;
    const int wm   = wave >> 1;   // 0..3
    const int wn   = wave & 1;    // 0..1

    const int rowBase = blockIdx.y * BM;
    const int colBase = blockIdx.x * BN;

    // ---- accumulators (C/D layout: VGPR r, lanes0-15 -> M=r, 16-31 -> M=r+8)
    v8f acc[2][4];
    for (int mt = 0; mt < 2; ++mt)
        for (int nt = 0; nt < 4; ++nt) {
            const int col = colBase + wn * 64 + nt * 16 + l15;
            for (int r = 0; r < 8; ++r) {
                float v = 0.0f;
                if (accumulate && col < N) {
                    const int row = rowBase + wm * 32 + mt * 16 + half * 8 + r;
                    v = C[(size_t)row * ldc + col];
                }
                acc[mt][nt][r] = v;
            }
        }

    // ---- tile loader helpers (each thread: 4 x float4 per tile)
    // element group e = tid + i*256 covers row r = e>>3, cols c4 = (e&7)*4
    const int nsteps = K / BK;

    // prologue: stage step 0 into buffer 0
    {
        #pragma unroll
        for (int i = 0; i < 4; ++i) {
            const int e = tid + i * 256;
            const int r = e >> 3, c4 = (e & 7) << 2;
            const float4 va = *(const float4*)&A[(size_t)(rowBase + r) * lda + c4];
            bf16x4 pa; pa[0]=(bf16_t)va.x; pa[1]=(bf16_t)va.y; pa[2]=(bf16_t)va.z; pa[3]=(bf16_t)va.w;
            *(bf16x4*)&As[0][r][c4] = pa;
            const int gr = colBase + r;
            float4 vb = make_float4(0.f, 0.f, 0.f, 0.f);
            if (gr < N) vb = *(const float4*)&W[(size_t)gr * ldb + c4];
            bf16x4 pb; pb[0]=(bf16_t)vb.x; pb[1]=(bf16_t)vb.y; pb[2]=(bf16_t)vb.z; pb[3]=(bf16_t)vb.w;
            *(bf16x4*)&Bs[0][r][c4] = pb;
        }
    }
    __syncthreads();

    for (int s = 0; s < nsteps; ++s) {
        const int cur = s & 1;
        const bool haveNext = (s + 1 < nsteps);
        const int k0n = (s + 1) * BK;

        // stage next tile in VGPRs (overlaps with WMMAs below)
        float4 stA[4], stB[4];
        if (haveNext) {
            #pragma unroll
            for (int i = 0; i < 4; ++i) {
                const int e = tid + i * 256;
                const int r = e >> 3, c4 = (e & 7) << 2;
                stA[i] = *(const float4*)&A[(size_t)(rowBase + r) * lda + k0n + c4];
                const int gr = colBase + r;
                stB[i] = make_float4(0.f, 0.f, 0.f, 0.f);
                if (gr < N) stB[i] = *(const float4*)&W[(size_t)gr * ldb + k0n + c4];
            }
        }

        // ---- fragments per ISA 7.12.2 16-bit layout:
        // lanes 0-15 hold K in {0..7, 16..23}; lanes 16-31 K in {8..15, 24..31}
        v16bf afrag[2], bfrag[4];
        #pragma unroll
        for (int mt = 0; mt < 2; ++mt) {
            const int row = wm * 32 + mt * 16 + l15;
            #pragma unroll
            for (int j = 0; j < 8; ++j) {
                const int kb = ((j < 4) ? (2 * j) : (2 * j + 8)) + half * 8;
                afrag[mt][2 * j]     = As[cur][row][kb];
                afrag[mt][2 * j + 1] = As[cur][row][kb + 1];
            }
        }
        #pragma unroll
        for (int nt = 0; nt < 4; ++nt) {
            const int col = wn * 64 + nt * 16 + l15;
            #pragma unroll
            for (int j = 0; j < 8; ++j) {
                const int kb = ((j < 4) ? (2 * j) : (2 * j + 8)) + half * 8;
                bfrag[nt][2 * j]     = Bs[cur][col][kb];
                bfrag[nt][2 * j + 1] = Bs[cur][col][kb + 1];
            }
        }
        #pragma unroll
        for (int mt = 0; mt < 2; ++mt)
            #pragma unroll
            for (int nt = 0; nt < 4; ++nt)
                acc[mt][nt] = __builtin_amdgcn_wmma_f32_16x16x32_bf16(
                    false, afrag[mt], false, bfrag[nt],
                    (short)0, acc[mt][nt], false, false);

        // write staged tile into the other buffer
        if (haveNext) {
            #pragma unroll
            for (int i = 0; i < 4; ++i) {
                const int e = tid + i * 256;
                const int r = e >> 3, c4 = (e & 7) << 2;
                bf16x4 pa; pa[0]=(bf16_t)stA[i].x; pa[1]=(bf16_t)stA[i].y;
                           pa[2]=(bf16_t)stA[i].z; pa[3]=(bf16_t)stA[i].w;
                *(bf16x4*)&As[cur ^ 1][r][c4] = pa;
                bf16x4 pb; pb[0]=(bf16_t)stB[i].x; pb[1]=(bf16_t)stB[i].y;
                           pb[2]=(bf16_t)stB[i].z; pb[3]=(bf16_t)stB[i].w;
                *(bf16x4*)&Bs[cur ^ 1][r][c4] = pb;
            }
        }
        __syncthreads();
    }

    // ---- epilogue store (rows always in range; only col guarded)
    for (int mt = 0; mt < 2; ++mt)
        for (int nt = 0; nt < 4; ++nt) {
            const int col = colBase + wn * 64 + nt * 16 + l15;
            if (col < N) {
                for (int r = 0; r < 8; ++r) {
                    const int row = rowBase + wm * 32 + mt * 16 + half * 8 + r;
                    float v = acc[mt][nt][r];
                    if (epilogue == 1)  // exact GELU
                        v = 0.5f * v * (1.0f + erff(v * 0.70710678118654752f));
                    C[(size_t)row * ldc + col] = v;
                }
            }
        }
}

// ---------------- elementwise / prep kernels ----------------

// At[row*NHEADS+h] = softplus(dt) * (-exp(A_log[h]))
__global__ void compute_at_kernel(const float* __restrict__ zx,
                                  const float* __restrict__ A_log,
                                  float* __restrict__ At, int total)
{
    const int i = blockIdx.x * blockDim.x + threadIdx.x;
    if (i >= total) return;
    const int row = i / NHEADS, h = i % NHEADS;
    const float dt = zx[(size_t)row * D_IN_PROJ + (D_INNER + CHANNELS) + h];
    const float sp = (dt <= 20.0f) ? log1pf(expf(dt)) : dt;
    At[i] = sp * (-expf(A_log[h]));
}

// xpad[b, t, ch] : 3 zero rows then xBC slice of zxbcdt
__global__ void pad_xbc_kernel(const float* __restrict__ zx,
                               float* __restrict__ xpad, size_t total)
{
    const size_t i = (size_t)blockIdx.x * blockDim.x + threadIdx.x;
    if (i >= total) return;
    const int ch = (int)(i % CHANNELS);
    const size_t r = i / CHANNELS;
    const int t = (int)(r % (LSEQ + 3));
    const int b = (int)(r / (LSEQ + 3));
    float v = 0.0f;
    if (t >= 3)
        v = zx[((size_t)b * LSEQ + (t - 3)) * D_IN_PROJ + D_INNER + ch];
    xpad[i] = v;
}

// wk[k][o][i] = conv_w[o][i][k]  (make each tap a contiguous [O,I] matrix)
__global__ void repack_convw_kernel(const float* __restrict__ cw,
                                    float* __restrict__ wk, size_t total)
{
    const size_t idx = (size_t)blockIdx.x * blockDim.x + threadIdx.x;
    if (idx >= total) return;
    const int ii = (int)(idx % CHANNELS);
    const size_t r = idx / CHANNELS;
    const int o = (int)(r % CHANNELS);
    const int k = (int)(r / CHANNELS);
    wk[idx] = cw[((size_t)o * CHANNELS + ii) * KSIZE + k];
}

// conv += bias; PReLU
__global__ void bias_prelu_kernel(float* __restrict__ conv,
                                  const float* __restrict__ cb,
                                  const float* __restrict__ pa, size_t total)
{
    const size_t i = (size_t)blockIdx.x * blockDim.x + threadIdx.x;
    if (i >= total) return;
    const int ch = (int)(i % CHANNELS);
    const float v = conv[i] + cb[ch];
    conv[i] = fmaxf(v, 0.0f) + pa[0] * fminf(v, 0.0f);
}

// ---------------- SSD as linear recurrence ----------------
// One block per (b,h). 256 threads keep the 64x64 state: p = tid&63,
// n-group = tid>>6 (16 n's each). y_t = C_t . S_t, S_t = e^{A_t} S + x B^T.
__global__ __launch_bounds__(256) void ssd_scan_kernel(
    const float* __restrict__ conv, const float* __restrict__ At,
    float* __restrict__ Y)
{
    __shared__ float xs[HEADDIM], Bsh[D_STATE], Csh[D_STATE], ash;
    __shared__ float red[4][HEADDIM];

    const int bh = blockIdx.x;
    const int b = bh / NHEADS, h = bh % NHEADS;
    const int tid = threadIdx.x;
    const int p  = tid & 63;
    const int ng = tid >> 6;

    float S[16];
    #pragma unroll
    for (int i = 0; i < 16; ++i) S[i] = 0.0f;

    for (int t = 0; t < LSEQ; ++t) {
        const size_t base = ((size_t)b * LSEQ + t) * CHANNELS;
        if (tid < 64)        xs[tid]        = conv[base + h * HEADDIM + tid];
        else if (tid < 128)  Bsh[tid - 64]  = conv[base + D_INNER + (tid - 64)];
        else if (tid < 192)  Csh[tid - 128] = conv[base + D_INNER + D_STATE + (tid - 128)];
        else if (tid == 192) ash = expf(At[((size_t)b * LSEQ + t) * NHEADS + h]);
        __syncthreads();

        const float a = ash, xp = xs[p];
        float part = 0.0f;
        #pragma unroll
        for (int i = 0; i < 16; ++i) {
            const int n = ng * 16 + i;
            S[i] = a * S[i] + xp * Bsh[n];
            part += Csh[n] * S[i];
        }
        red[ng][p] = part;
        __syncthreads();

        if (tid < 64) {
            const float y = red[0][tid] + red[1][tid] + red[2][tid] + red[3][tid];
            Y[((size_t)b * LSEQ + t) * D_INNER + h * HEADDIM + tid] = y;
        }
        // next iteration's first __syncthreads() fences red/xs reuse
    }
}

// ---------------- h = u2*Y ; LayerNorm(2048) in place over u2 ----------------
__global__ __launch_bounds__(256) void lnmul_kernel(
    float* __restrict__ u2, const float* __restrict__ Y,
    const float* __restrict__ g, const float* __restrict__ bta)
{
    __shared__ float red[256];
    const int row = blockIdx.x;
    const int tid = threadIdx.x;
    const float* yr = Y + (size_t)row * D_INNER;
    float* hr = u2 + (size_t)row * D_INNER;

    float local[8];
    float s = 0.0f;
    #pragma unroll
    for (int i = 0; i < 8; ++i) {
        const int c = tid + i * 256;
        const float h = hr[c] * yr[c];
        local[i] = h;
        s += h;
    }
    red[tid] = s; __syncthreads();
    for (int off = 128; off > 0; off >>= 1) {
        if (tid < off) red[tid] += red[tid + off];
        __syncthreads();
    }
    const float mu = red[0] * (1.0f / D_INNER);
    __syncthreads();

    float v = 0.0f;
    #pragma unroll
    for (int i = 0; i < 8; ++i) { const float d = local[i] - mu; v += d * d; }
    red[tid] = v; __syncthreads();
    for (int off = 128; off > 0; off >>= 1) {
        if (tid < off) red[tid] += red[tid + off];
        __syncthreads();
    }
    const float inv = rsqrtf(red[0] * (1.0f / D_INNER) + 1e-5f);
    #pragma unroll
    for (int i = 0; i < 8; ++i) {
        const int c = tid + i * 256;
        hr[c] = (local[i] - mu) * inv * g[c] + bta[c];
    }
}

// ---------------- launch ----------------
extern "C" void kernel_launch(void* const* d_in, const int* in_sizes, int n_in,
                              void* d_out, int out_size, void* d_ws, size_t ws_size,
                              hipStream_t stream)
{
    const float* u       = (const float*)d_in[0];
    const float* W_in1   = (const float*)d_in[1];
    const float* A_log   = (const float*)d_in[2];
    const float* conv_w  = (const float*)d_in[3];
    const float* conv_b  = (const float*)d_in[4];
    const float* prelu_a = (const float*)d_in[5];
    const float* W_in2   = (const float*)d_in[6];
    const float* W_in3   = (const float*)d_in[7];
    const float* ln_g    = (const float*)d_in[8];
    const float* ln_b    = (const float*)d_in[9];
    float* out = (float*)d_out;

    float* ws = (float*)d_ws;
    size_t off = 0;
    float* zx   = ws + off; off += (size_t)BSZ * LSEQ * D_IN_PROJ;          // 4096x4256
    float* xpad = ws + off; off += (size_t)BSZ * (LSEQ + 3) * CHANNELS;     // padded xBC
    float* wk   = ws + off; off += (size_t)KSIZE * CHANNELS * CHANNELS;     // tap matrices
    float* conv = ws + off; off += (size_t)BSZ * LSEQ * CHANNELS;           // conv out
    float* At   = ws + off; off += (size_t)BSZ * LSEQ * NHEADS;             // dt*A
    float* Y    = ws + off; off += (size_t)BSZ * LSEQ * D_INNER;            // SSD out
    float* u2   = ws + off; off += (size_t)BSZ * LSEQ * D_INNER;            // gelu branch / hn

    const int M = BSZ * LSEQ; // 4096 rows

    // 1) zxbcdt = u @ W_in1^T
    {
        dim3 g((D_IN_PROJ + BN - 1) / BN, (M + BM - 1) / BM);
        gemm_bf16_wmma<<<g, 256, 0, stream>>>(u, D_MODEL, W_in1, D_MODEL,
                                              zx, D_IN_PROJ, M, D_IN_PROJ, D_MODEL, 0, 0);
    }
    // 2) At = softplus(dt) * (-exp(A_log))
    {
        const int total = M * NHEADS;
        compute_at_kernel<<<(total + 255) / 256, 256, 0, stream>>>(zx, A_log, At, total);
    }
    // 3) causal-pad xBC
    {
        const size_t total = (size_t)BSZ * (LSEQ + 3) * CHANNELS;
        pad_xbc_kernel<<<(int)((total + 255) / 256), 256, 0, stream>>>(zx, xpad, total);
    }
    // 4) repack conv_w into per-tap [O,I] matrices
    {
        const size_t total = (size_t)KSIZE * CHANNELS * CHANNELS;
        repack_convw_kernel<<<(int)((total + 255) / 256), 256, 0, stream>>>(conv_w, wk, total);
    }
    // 5) full conv = 4 accumulated GEMMs per batch (shifted padded rows)
    for (int b = 0; b < BSZ; ++b)
        for (int k = 0; k < KSIZE; ++k) {
            dim3 g((CHANNELS + BN - 1) / BN, (LSEQ + BM - 1) / BM);
            const float* Ak = xpad + (size_t)b * (LSEQ + 3) * CHANNELS + (size_t)k * CHANNELS;
            gemm_bf16_wmma<<<g, 256, 0, stream>>>(Ak, CHANNELS,
                                                  wk + (size_t)k * CHANNELS * CHANNELS, CHANNELS,
                                                  conv + (size_t)b * LSEQ * CHANNELS, CHANNELS,
                                                  LSEQ, CHANNELS, CHANNELS, (k > 0) ? 1 : 0, 0);
        }
    // 6) bias + PReLU
    {
        const size_t total = (size_t)BSZ * LSEQ * CHANNELS;
        bias_prelu_kernel<<<(int)((total + 255) / 256), 256, 0, stream>>>(conv, conv_b, prelu_a, total);
    }
    // 7) SSD linear recurrence -> Y
    ssd_scan_kernel<<<BSZ * NHEADS, 256, 0, stream>>>(conv, At, Y);
    // 8) u2 = gelu(u @ W_in2^T)
    {
        dim3 g((D_INNER + BN - 1) / BN, (M + BM - 1) / BM);
        gemm_bf16_wmma<<<g, 256, 0, stream>>>(u, D_MODEL, W_in2, D_MODEL,
                                              u2, D_INNER, M, D_INNER, D_MODEL, 0, 1);
    }
    // 9) h = u2*Y ; LayerNorm -> u2
    lnmul_kernel<<<M, 256, 0, stream>>>(u2, Y, ln_g, ln_b);
    // 10) out = hn @ W_in3^T
    {
        dim3 g((D_MODEL + BN - 1) / BN, (M + BM - 1) / BM);
        gemm_bf16_wmma<<<g, 256, 0, stream>>>(u2, D_INNER, W_in3, D_INNER,
                                              out, D_MODEL, M, D_MODEL, D_INNER, 0, 0);
    }
}
